// ConvSE3_42099269435999
// MI455X (gfx1250) — matrix operand order repaired and checked by reading, not compiled, not run on hardware
//
#include <hip/hip_runtime.h>
#include <math.h>

// ---------------------------------------------------------------------------
// SE(3) conv for MI455X (gfx1250, wave32, v_wmma_f32_16x16x32_f16).
// Grid = 128 node-groups x 4 pairs (512 blocks, 512 threads = 16 waves).
// Each block: 4 nodes x 32 edges = 128 edge rows through one pair's radial
// MLP (WMMA GEMMs + fused LayerNorm/GELU), then pooling is commuted in front
// of the w3 projection:
//   out[mo,d] = sum_{c,r} w3[c, mo*Mnf+r] * Q[c,r,d],
//   Q[c,r,d]  = sum_e h2[e,c] * tmp[e,r,d] * mask[e]
// Packing 4 nodes into the N dimension of the final long-K GEMM raises its
// column utilization from dout/16 to 4*dout/16 and cuts w3 L2 traffic 4x.
// c is processed in 4 chunks of 32 so the packed Q chunk fits in LDS.
// Cross-wave/cross-pair reduction via global_atomic_add_f32 into zeroed out.
// ---------------------------------------------------------------------------

typedef _Float16 half16 __attribute__((ext_vector_type(16)));
typedef float    float8 __attribute__((ext_vector_type(8)));

#define NNODES 512
#define MIDD   128

// LDS byte offsets (dynamic shared memory, base offset 0: no static LDS here)
#define OFF_TT   0         // [288][128] f16  tmp^T (masked)        73728 B
#define OFF_H    73728     // [128][128] f16  hidden                 32768 B
#define OFF_HT   106496    // [128][128] f16  hidden^T (h2 only)     32768 B
#define OFF_E    139264    // [128][64]  f16  edges                  16384 B
#define OFF_R    155648    // shared region: sel f32 | sG f32 | QP   73728 B
#define OFF_RED  229376    // [128][4][2] f32 LN partials             4096 B
#define OFF_MASK 233472    // [128] f32                                512 B
#define OFF_DEN  233984    // [4]   f32                                 16 B
#define LDS_TOTAL 234496

struct SE3Params {
  const float* feats0;
  const float* feats1;
  const float* edges;
  const float* basis[4];
  const int* nbr_idx;
  const unsigned char* nbr_mask;
  const float* b1[4]; const float* g1[4]; const float* be1[4];
  const float* b2[4]; const float* g2[4]; const float* be2[4];
  const _Float16* w1t[4];   // [128][64]  = w1^T, f16
  const _Float16* w2t[4];   // [128][128] = w2^T, f16
  const _Float16* w3a[4];   // [32][128*Mnf]: w3a[mo][c*Mnf+r] = w3[c][mo*Mnf+r]
  float* out0;
  float* out1;
};

struct PrepParams {
  const float* w1[4];
  const float* w2[4];
  const float* w3[4];
  _Float16* ws;
};

// ---- CDNA5 async global->LDS (per-lane scattered gather) -------------------
__device__ inline void async_ld_b32(unsigned lds_byte_off, const float* src) {
  asm volatile("global_load_async_to_lds_b32 %0, %1, off"
               :: "v"(lds_byte_off), "v"(src) : "memory");
}
__device__ inline void wait_async0() {
  asm volatile("s_wait_asynccnt 0" ::: "memory");
}

// ---- WMMA fragment helpers (A/B: contiguous k = 16*(l>>4) + s; C/D per CDNA5 ISA 7.12.2) ------------------
__device__ inline float8 wmma32(half16 a, half16 b, float8 c) {
  return __builtin_amdgcn_wmma_f32_16x16x32_f16(false, a, false, b, (short)0, c,
                                                false, false);
}

// A fragment 16x32 from row-major [m][k]: lane l holds M=l%16,
// VGPR v halves hold K = (l>>4)*16 + 2v + {0,1}.
__device__ inline half16 frag_a(const _Float16* A, int m0, int lda, int k0, int lane) {
  const int ml = lane & 15, g = lane >> 4;
  const _Float16* row = A + (m0 + ml) * lda + k0 + g * 16;
  half16 out;
#pragma unroll
  for (int v = 0; v < 8; ++v) {
    const int kb = 2 * v;
    out[2 * v]     = row[kb];
    out[2 * v + 1] = row[kb + 1];
  }
  return out;
}

// B fragment 32x16 from TRANSPOSED row-major storage BT[n][k]:
// lane l holds N=l%16, VGPR v halves hold K = (l>>4)*16 + 2v + {0,1}.
__device__ inline half16 frag_bt(const _Float16* BT, int n0, int ldb, int k0, int lane) {
  const int nl = lane & 15, g = lane >> 4;
  const _Float16* row = BT + (n0 + nl) * ldb + k0 + g * 16;
  half16 out;
#pragma unroll
  for (int v = 0; v < 8; ++v) {
    out[2 * v]     = row[2 * v];
    out[2 * v + 1] = row[2 * v + 1];
  }
  return out;
}

// B fragment from packed Q chunk: QP is k-linear [k_chunk][D4] f16,
// B[k][n] = (n < D4valid) ? QP[k*D4 + n] : 0.
template <int D4>
__device__ inline half16 frag_bqp(const _Float16* QP, int kc0, int lane) {
  const int nl = lane & 15, g = lane >> 4;
  half16 out;
  if (nl < D4) {
    const _Float16* q = QP + (kc0 + g * 16) * D4 + nl;
#pragma unroll
    for (int v = 0; v < 8; ++v) {
      out[2 * v]     = q[(2 * v) * D4];
      out[2 * v + 1] = q[(2 * v + 1) * D4];
    }
  } else {
#pragma unroll
    for (int v = 0; v < 8; ++v) {
      out[2 * v]     = (_Float16)0.f;
      out[2 * v + 1] = (_Float16)0.f;
    }
  }
  return out;
}

// C/D fragment: VGPR r -> M = m0 + r + 8*(l>>4), N = n0 + (l&15).
__device__ inline void frag_store(float* D, int m0, int ldd, int n0, int lane, float8 d) {
  const int nl = lane & 15, g = lane >> 4;
#pragma unroll
  for (int r = 0; r < 8; ++r) D[(m0 + r + 8 * g) * ldd + n0 + nl] = d[r];
}

// LayerNorm(+bias) + exact GELU: 128 rows x 128 cols, 512 threads.
__device__ inline void ln_gelu(const float* sG, const float* bias, const float* gamma,
                               const float* beta, _Float16* sH, _Float16* sHT,
                               float* sRed, int t, bool writeT) {
  __syncthreads();  // sG ready
  const int row = t >> 2, ch = t & 3;  // 128 rows x 4 chunks of 32 cols
  const float* src = sG + row * MIDD + ch * 32;
  float s = 0.f, s2 = 0.f;
#pragma unroll
  for (int c = 0; c < 32; ++c) {
    float x = src[c] + bias[ch * 32 + c];
    s += x; s2 += x * x;
  }
  sRed[(row * 4 + ch) * 2]     = s;
  sRed[(row * 4 + ch) * 2 + 1] = s2;
  __syncthreads();
  float S = 0.f, S2 = 0.f;
#pragma unroll
  for (int j = 0; j < 4; ++j) {
    S  += sRed[(row * 4 + j) * 2];
    S2 += sRed[(row * 4 + j) * 2 + 1];
  }
  const float mean = S * (1.0f / MIDD);
  const float var  = S2 * (1.0f / MIDD) - mean * mean;
  const float inv  = rsqrtf(var + 1e-5f);
#pragma unroll
  for (int c = 0; c < 32; ++c) {
    const int col = ch * 32 + c;
    float x = src[c] + bias[col];
    float y = (x - mean) * inv * gamma[col] + beta[col];
    float a = 0.5f * y * (1.0f + erff(y * 0.70710678118654752f));
    _Float16 h = (_Float16)a;
    sH[row * MIDD + col] = h;
    if (writeT) sHT[col * MIDD + row] = h;
  }
  __syncthreads();
}

// ---- weight prep: f32 -> f16, transposed / rearranged ----------------------
__global__ __launch_bounds__(256) void se3_prep_kernel(PrepParams pp) {
  const int idx = blockIdx.x * 256 + threadIdx.x;
  _Float16* ws = pp.ws;
  if (idx < 32768) {                       // w1^T: 4 x [128][64]
    const int p = idx >> 13, r = idx & 8191;
    const int o = r >> 6, i = r & 63;
    ws[idx] = (_Float16)pp.w1[p][i * 128 + o];
  } else if (idx < 98304) {                // w2^T: 4 x [128][128]
    const int l = idx - 32768;
    const int p = l >> 14, r = l & 16383;
    const int o = r >> 7, i = r & 127;
    ws[idx] = (_Float16)pp.w2[p][i * 128 + o];
  } else if (idx < 884736) {               // w3a: [mo][c*Mnf+r] = w3[c][mo*Mnf+r]
    const int l = idx - 98304;
    int p, loc, K, Mnf;
    if (l < 393216) { p = l >> 17; loc = l & 131071; K = 4096;  Mnf = 32; }
    else            { p = 3;       loc = l - 393216; K = 12288; Mnf = 96; }
    const int mo = loc / K, k = loc - mo * K;
    const int c = k / Mnf, r = k - c * Mnf;
    ws[idx] = (_Float16)pp.w3[p][c * (32 * Mnf) + mo * Mnf + r];
  }
}

__global__ __launch_bounds__(256) void se3_zero_kernel(float* o, int n) {
  const int i = blockIdx.x * 256 + threadIdx.x;
  if (i < n) o[i] = 0.f;
}

// ---- main body: one block = 4 nodes x 1 pair -------------------------------
template <int PAIR>
__device__ void se3_body(const SE3Params& P, int grp, char* smem) {
  constexpr int Mnf  = (PAIR == 3) ? 96 : 32;
  constexpr int dout = (PAIR == 1 || PAIR == 3) ? 3 : 1;
  constexpr int RD   = Mnf * dout;          // 32 | 96 | 32 | 288
  constexpr int NT   = RD / 16;             // 2 | 6 | 2 | 18
  constexpr int KT   = MIDD * Mnf;          // 4096 | 12288
  constexpr int D4   = 4 * dout;            // packed N columns (4 nodes)

  _Float16* sTT = (_Float16*)(smem + OFF_TT);   // [RD][128]
  _Float16* sH  = (_Float16*)(smem + OFF_H);    // [128][128]
  _Float16* sHT = (_Float16*)(smem + OFF_HT);   // [128][128] (h2^T)
  _Float16* sE  = (_Float16*)(smem + OFF_E);    // [128][64]
  float*    sSel= (float*)   (smem + OFF_R);    // sel0 [128][32] | sel1 [128][96]
  float*    sG  = (float*)   (smem + OFF_R);    // [128][128] GEMM f32 scratch
  _Float16* sQP = (_Float16*)(smem + OFF_R);    // [32*Mnf][D4] packed Q chunk
  float*    sRed= (float*)   (smem + OFF_RED);
  float*    sMask=(float*)   (smem + OFF_MASK); // [128]
  float*    sDen= (float*)   (smem + OFF_DEN);  // [4]

  const int t = threadIdx.x;
  const int lane = t & 31;
  const int wave = t >> 5;                  // 16 waves
  const int eg0 = grp * 128;                // first global edge of this group
  const int n0  = grp * 4;                  // first global node

  // ---- stage: edges (f32->f16), mask, async gather of neighbor feats ------
  for (int i = t; i < 128 * 64; i += 512) sE[i] = (_Float16)P.edges[eg0 * 64 + i];
  if (t < 128) sMask[t] = P.nbr_mask[eg0 + t] ? 1.0f : 0.0f;
  if (PAIR <= 1) {  // needs feats0 gather: [128 edges][32]
    for (int i = t; i < 128 * 32; i += 512) {
      const int e = i >> 5, m = i & 31;
      async_ld_b32(OFF_R + i * 4, P.feats0 + P.nbr_idx[eg0 + e] * 32 + m);
    }
  } else {          // needs feats1 gather: [128 edges][96]
    for (int i = t; i < 128 * 96; i += 512) {
      const int e = i / 96, md = i - e * 96;
      async_ld_b32(OFF_R + i * 4, P.feats1 + P.nbr_idx[eg0 + e] * 96 + md);
    }
  }
  wait_async0();
  __syncthreads();
  if (t < 4) {
    float s = 0.f;
    for (int e = 0; e < 32; ++e) s += sMask[t * 32 + e];
    sDen[t] = fmaxf(s, 1.0f);
  }

  // ---- tmp[e,r,d] = (feats@basis)*mask, stored transposed [rd][e] ---------
  const float* basis = P.basis[PAIR];
  for (int i = t; i < RD * 128; i += 512) {
    const int rd = i >> 7, e = i & 127;
    const int eg = eg0 + e;
    float val;
    if (PAIR == 0) {
      val = sSel[e * 32 + rd] * basis[eg];
    } else if (PAIR == 1) {
      const int r = rd / 3, d = rd - 3 * r;
      val = sSel[e * 32 + r] * basis[eg * 3 + d];
    } else if (PAIR == 2) {
      const float* b = basis + eg * 3;
      const float* s1 = sSel + e * 96 + rd * 3;
      val = s1[0] * b[0] + s1[1] * b[1] + s1[2] * b[2];
    } else {
      const int q = rd / 3, d = rd - 3 * q;
      const int mi = q / 3, f = q - 3 * mi;
      const float* b = basis + eg * 27 + f * 3 + d;
      const float* s1 = sSel + e * 96 + mi * 3;
      val = s1[0] * b[0] + s1[1] * b[9] + s1[2] * b[18];
    }
    sTT[rd * 128 + e] = (_Float16)(val * sMask[e]);
  }
  __syncthreads();  // sSel dead; region R becomes sG

  // ---- GEMM1: [128x64] @ w1[64x128] -> sG (8x8 tiles, 2 K-steps) ----------
  for (int tid = wave; tid < 64; tid += 16) {
    const int mt = tid & 7, nt = tid >> 3;
    float8 acc = {};
#pragma unroll
    for (int kk = 0; kk < 64; kk += 32) {
      half16 a = frag_a(sE, mt * 16, 64, kk, lane);
      half16 b = frag_bt(P.w1t[PAIR], nt * 16, 64, kk, lane);
      acc = wmma32(a, b, acc);
    }
    frag_store(sG, mt * 16, MIDD, nt * 16, lane, acc);
  }
  ln_gelu(sG, P.b1[PAIR], P.g1[PAIR], P.be1[PAIR], sH, sHT, sRed, t, false);

  // ---- GEMM2: [128x128] @ w2[128x128] -> sG -------------------------------
  for (int tid = wave; tid < 64; tid += 16) {
    const int mt = tid & 7, nt = tid >> 3;
    float8 acc = {};
#pragma unroll
    for (int kk = 0; kk < 128; kk += 32) {
      half16 a = frag_a(sH, mt * 16, MIDD, kk, lane);
      half16 b = frag_bt(P.w2t[PAIR], nt * 16, MIDD, kk, lane);
      acc = wmma32(a, b, acc);
    }
    frag_store(sG, mt * 16, MIDD, nt * 16, lane, acc);
  }
  ln_gelu(sG, P.b2[PAIR], P.g2[PAIR], P.be2[PAIR], sH, sHT, sRed, t, true);
  // region R now becomes sQP (LN exit barrier fenced all sG reads)

  // ---- c-chunk loop: Q-GEMM into packed chunk, then partial final GEMM ----
  const _Float16* w3a = P.w3a[PAIR];
  float8 acc0 = {}, acc1 = {};
#pragma unroll 1
  for (int chunk = 0; chunk < 4; ++chunk) {
    const int c0 = chunk * 32;
    // Q[c, r, d] = sum_e h2[e,c]*tmp[e,rd] per node; K=32 = one WMMA.
    // Tiles: node(4) x ct(2) x rt(NT); counts 16|48|16|144, all %16==0.
    for (int tid = wave; tid < 8 * NT; tid += 16) {
      const int nd = tid / (2 * NT);
      const int rem = tid - nd * 2 * NT;
      const int ct = rem / NT, rt = rem - ct * NT;
      half16 a = frag_a(sHT, c0 + ct * 16, MIDD, nd * 32, lane);
      half16 b = frag_bt(sTT, rt * 16, 128, nd * 32, lane);
      float8 acc = {};
      acc = wmma32(a, b, acc);
      const int nl = lane & 15, gl = lane >> 4;
#pragma unroll
      for (int r = 0; r < 8; ++r) {
        const int cl = ct * 16 + r + 8 * gl;        // local c in chunk
        const int rdl = rt * 16 + nl;               // [0, RD)
        const int rr = rdl / dout, dd = rdl - rr * dout;
        sQP[(cl * Mnf + rr) * D4 + nd * dout + dd] = (_Float16)acc[r];
      }
    }
    __syncthreads();

    // out[mo, n] += sum_{k in chunk} w3a[mo, c0*Mnf + k] * QP[k, n]
    for (int s = wave; s < Mnf; s += 16) {          // Mnf/16 steps per wave
      const int kc = s * 32;
      half16 b  = frag_bqp<D4>(sQP, kc, lane);
      half16 a0 = frag_a(w3a, 0, KT, c0 * Mnf + kc, lane);
      half16 a1 = frag_a(w3a, 16, KT, c0 * Mnf + kc, lane);
      acc0 = wmma32(a0, b, acc0);
      acc1 = wmma32(a1, b, acc1);
      __builtin_prefetch((const void*)(w3a + (lane & 15) * KT + c0 * Mnf + kc + 512),
                         0, 1);
    }
    __syncthreads();  // sQP rewritten next chunk
  }

  // ---- epilogue: pre-scaled global atomic accumulation --------------------
  const int nl = lane & 15, gl = lane >> 4;
  if (nl < D4) {
    const int nd = nl / dout, dd = nl - nd * dout;
    const float invd = 1.0f / sDen[nd];
    const int gn = n0 + nd;
    if (dout == 1) {
#pragma unroll
      for (int r = 0; r < 8; ++r) {
        atomicAdd(P.out0 + gn * 32 + (r + 8 * gl), acc0[r] * invd);
        atomicAdd(P.out0 + gn * 32 + (16 + r + 8 * gl), acc1[r] * invd);
      }
    } else {
#pragma unroll
      for (int r = 0; r < 8; ++r) {
        atomicAdd(P.out1 + gn * 96 + (r + 8 * gl) * 3 + dd, acc0[r] * invd);
        atomicAdd(P.out1 + gn * 96 + (16 + r + 8 * gl) * 3 + dd, acc1[r] * invd);
      }
    }
  }
}

__global__ __launch_bounds__(512) void se3_main_kernel(SE3Params P) {
  extern __shared__ char smem_raw[];
  const int p = blockIdx.x & 3, grp = blockIdx.x >> 2;
  switch (p) {
    case 0: se3_body<0>(P, grp, smem_raw); break;
    case 1: se3_body<1>(P, grp, smem_raw); break;
    case 2: se3_body<2>(P, grp, smem_raw); break;
    default: se3_body<3>(P, grp, smem_raw); break;
  }
}

// ---------------------------------------------------------------------------
extern "C" void kernel_launch(void* const* d_in, const int* in_sizes, int n_in,
                              void* d_out, int out_size, void* d_ws, size_t ws_size,
                              hipStream_t stream) {
  (void)in_sizes; (void)n_in; (void)ws_size;
  _Float16* ws = (_Float16*)d_ws;

  se3_zero_kernel<<<256, 256, 0, stream>>>((float*)d_out, out_size);

  PrepParams pp;
  for (int p = 0; p < 4; ++p) {
    const int base = 9 + p * 9;
    pp.w1[p] = (const float*)d_in[base + 0];
    pp.w2[p] = (const float*)d_in[base + 4];
    pp.w3[p] = (const float*)d_in[base + 8];
  }
  pp.ws = ws;
  se3_prep_kernel<<<3456, 256, 0, stream>>>(pp);  // 884736 f16 weight elems

  SE3Params P;
  P.feats0 = (const float*)d_in[0];
  P.feats1 = (const float*)d_in[1];
  P.edges  = (const float*)d_in[2];
  P.basis[0] = (const float*)d_in[3];
  P.basis[1] = (const float*)d_in[4];
  P.basis[2] = (const float*)d_in[5];
  P.basis[3] = (const float*)d_in[6];
  P.nbr_idx  = (const int*)d_in[7];
  P.nbr_mask = (const unsigned char*)d_in[8];
  static const int w3off[4] = {98304, 229376, 360448, 491520};
  for (int p = 0; p < 4; ++p) {
    const int base = 9 + p * 9;
    P.b1[p]  = (const float*)d_in[base + 1];
    P.g1[p]  = (const float*)d_in[base + 2];
    P.be1[p] = (const float*)d_in[base + 3];
    P.b2[p]  = (const float*)d_in[base + 5];
    P.g2[p]  = (const float*)d_in[base + 6];
    P.be2[p] = (const float*)d_in[base + 7];
    P.w1t[p] = ws + p * 8192;
    P.w2t[p] = ws + 32768 + p * 16384;
    P.w3a[p] = ws + w3off[p];
  }
  P.out0 = (float*)d_out;
  P.out1 = (float*)d_out + NNODES * 32;

  hipFuncSetAttribute((const void*)se3_main_kernel,
                      hipFuncAttributeMaxDynamicSharedMemorySize, LDS_TOTAL);
  se3_main_kernel<<<NNODES, 512, LDS_TOTAL, stream>>>(P);
}
